// RPN_146028888278
// MI455X (gfx1250) — matrix operand order repaired and checked by reading, not compiled, not run on hardware
//
#include <hip/hip_runtime.h>
#include <hip/hip_bf16.h>

// ---------------------------------------------------------------------------
// Types for CDNA5 WMMA (wave32, 16x16x32 bf16 -> f32) and TDM descriptors
// ---------------------------------------------------------------------------
typedef __attribute__((ext_vector_type(16))) __bf16 v16bf;
typedef __attribute__((ext_vector_type(8)))  __bf16 v8bf;
typedef __attribute__((ext_vector_type(8)))  float  v8f;
typedef __attribute__((ext_vector_type(4)))  unsigned int u32x4;
typedef __attribute__((ext_vector_type(8)))  int i32x8;
typedef __attribute__((ext_vector_type(4)))  int i32x4;

static_assert(sizeof(v16bf) == 32, "v16bf must be 32 bytes");
static_assert(sizeof(v8bf) == 16, "v8bf must be 16 bytes");

#define IMG_WH   1024.0f
#define PRE_NMS  2000
#define POST_NMS 1000
#define IOU_THR  0.7f
#define MIN_SIZE 1.0f
#define PROB_THR 0.05f

// Problem dims
#define NB   8          // batch
#define CIN  256
#define COUT 256
#define HH   64
#define WW   64
#define HP   66         // padded
#define AN   9          // anchors
#define NPROP (AN*HH*WW)    // 36864 proposals per batch
#define SORTN 65536          // padded sort length
#define KCONV (9*CIN)        // 2304
#define NHEAD 48             // 36 box + 9 obj + 3 pad

// Workspace layout (bytes, 256-aligned)
#define OFF_FEATPAD  0ull
#define OFF_CONVW    17842176ull
#define OFF_HEADW    19021824ull
#define OFF_CONVOUT  19046400ull
#define OFF_HEADOUT  35823616ull
#define OFF_BOXES    42115072ull
#define OFF_PROB     46833664ull
#define OFF_KEYS     48013312ull
#define OFF_TOPBOX   52207616ull
#define OFF_KEEP     52463616ull

// TDM availability (compile-probe guarded; fallback keeps Round-1 code)
#if defined(__HIP_DEVICE_COMPILE__) && \
    __has_builtin(__builtin_amdgcn_tensor_load_to_lds) && \
    __has_builtin(__builtin_amdgcn_s_wait_tensorcnt)
#define RPN_HAS_TDM 1
#else
#define RPN_HAS_TDM 0
#endif

// ---------------------------------------------------------------------------
// Stage 1a: NCHW f32 -> padded NHWC bf16
// ---------------------------------------------------------------------------
__global__ void rpn_prep_feats(const float* __restrict__ feat,
                               __bf16* __restrict__ featpad) {
    int gid = blockIdx.x * blockDim.x + threadIdx.x;  // 8*66*66*256
    int c  = gid & (CIN - 1);
    int p  = gid >> 8;
    int wp = p % HP; int t = p / HP;
    int hp = t % HP; int n = t / HP;
    float v = 0.0f;
    int h = hp - 1, w = wp - 1;
    if (h >= 0 && h < HH && w >= 0 && w < WW)
        v = feat[(((size_t)n * CIN + c) * HH + h) * WW + w];
    featpad[gid] = (__bf16)v;
}

// Stage 1b: conv weights (cout, cin, 3, 3) f32 -> Bt[cout][k=rs*256+c] bf16
__global__ void rpn_prep_convw(const float* __restrict__ w1,
                               __bf16* __restrict__ convW) {
    int gid = blockIdx.x * blockDim.x + threadIdx.x;  // 256*2304
    int k    = gid % KCONV;
    int cout = gid / KCONV;
    int c  = k & (CIN - 1);
    int rs = k >> 8;          // 0..8, (r*3+s)
    convW[gid] = (__bf16)w1[(size_t)cout * KCONV + c * 9 + rs];
}

// Stage 1c: head weights -> Bt[o][c] bf16 (rows 0..35 box, 36..44 obj, pad 0)
__global__ void rpn_prep_headw(const float* __restrict__ w_box,
                               const float* __restrict__ w_obj,
                               __bf16* __restrict__ headW) {
    int gid = blockIdx.x * blockDim.x + threadIdx.x;  // 48*256
    int c = gid & (CIN - 1);
    int o = gid >> 8;
    float v = 0.0f;
    if (o < 36)      v = w_box[o * CIN + c];
    else if (o < 45) v = w_obj[(o - 36) * CIN + c];
    headW[gid] = (__bf16)v;
}

// ---------------------------------------------------------------------------
// A-fragment loader: lane holds row m, K-halves split at lane 16
// (lanes 0-15: K 0-15; lanes 16-31: K 16-31)
// ---------------------------------------------------------------------------
__device__ __forceinline__ v16bf load_afrag(const __bf16* p, int khalf) {
    v8bf lo = *(const v8bf*)(p + 2 * khalf);
    v8bf hi = *(const v8bf*)(p + 2 * khalf + 8);
    return __builtin_shufflevector(lo, hi, 0, 1, 2, 3, 4, 5, 6, 7,
                                           8, 9, 10, 11, 12, 13, 14, 15);
}

#if RPN_HAS_TDM
// ---------------------------------------------------------------------------
// TDM: async-load a 2D weight tile (64 cols x 32 k, bf16) from Bt[col][k]
// into LDS.  D# per CDNA5 ISA:
//   g0: count=1 | lds_addr | global_addr[56:0] | type=2
//   g1: data_size=2B, tensor_dim0=2304, tensor_dim1=256,
//       tile_dim0=32, tile_dim1=64, tensor_dim0_stride=2304
//   g2/g3: zero (2D tensor)
// ---------------------------------------------------------------------------
__device__ __forceinline__ void rpn_tdm_load_btile(unsigned lds_addr,
                                                   unsigned long long gaddr) {
    u32x4 g0 = { 1u,                                           // count=1
                 lds_addr,                                     // LDS byte addr
                 (unsigned)(gaddr & 0xFFFFFFFFull),            // gaddr[95:64]
                 (unsigned)((gaddr >> 32) & 0x01FFFFFFull) |
                     (2u << 30) };                             // addr hi | type=2
    i32x8 g1 = { (int)(1u << 16),          // workgroup_mask=0, data_size=2B
                 (int)(KCONV << 16),       // tensor_dim0 lo16 (bits 63:48)
                 (int)(256 << 16),         // dim0 hi16=0, tensor_dim1 lo16
                 (int)(32 << 16),          // dim1 hi16=0, tile_dim0=32
                 64,                       // tile_dim1=64, tile_dim2=0
                 KCONV,                    // tensor_dim0_stride lo32
                 0, 0 };                   // stride hi, dim1_stride (unused)
    i32x4 gz = { 0, 0, 0, 0 };
#if __clang_major__ >= 23
    i32x8 gz8 = { 0, 0, 0, 0, 0, 0, 0, 0 };
    __builtin_amdgcn_tensor_load_to_lds(g0, g1, gz, gz, gz8, 0);
#else
    __builtin_amdgcn_tensor_load_to_lds(g0, g1, gz, gz, 0);
#endif
}
#endif  // RPN_HAS_TDM

// ---------------------------------------------------------------------------
// Stage 2: 3x3 conv as implicit GEMM.
// Per batch: O[4096][256] = Im2col[4096][2304] x W[2304][256].
// Workgroup 256 thr = 8 waves; tile 128M x 64N; wave tile 32x32 (2x2 frags).
// Weights staged in LDS via TDM (double-buffered, TENSORcnt-tracked);
// activations streamed from global with prefetch.
// grid = (32 Mtiles, 4 Ntiles, 8 batches)
// ---------------------------------------------------------------------------
__global__ __launch_bounds__(256)
void rpn_conv_wmma(const __bf16* __restrict__ featpad,
                   const __bf16* __restrict__ convW,
                   const float* __restrict__ b1,
                   __bf16* __restrict__ convout) {
#if RPN_HAS_TDM
    __shared__ __bf16 sB[2][64 * 32];     // 4KB per buffer, double buffered
#endif
    const int n    = blockIdx.z;
    const int m0wg = blockIdx.x * 128;
    const int n0wg = blockIdx.y * 64;
    const int wave = threadIdx.x >> 5;
    const int lane = threadIdx.x & 31;
    const int mw = wave >> 1;            // 0..3  (M direction)
    const int nw = wave & 1;             // 0..1  (N direction)
    const int m0 = m0wg + mw * 32;
    const int n0 = n0wg + nw * 32;

    const int laneM = lane & 15;
    const int khalf = (lane < 16) ? 0 : 8;    // A split
    const int kb16  = (lane < 16) ? 0 : 16;   // B split

    // A rows (pixels) for the two M-fragments
    const int rowA0 = m0 + laneM;
    const int rowA1 = m0 + 16 + laneM;
    const int h0 = rowA0 >> 6, w0 = rowA0 & 63;
    const int h1 = rowA1 >> 6, w1v = rowA1 & 63;

#if RPN_HAS_TDM
    const unsigned ldsBase = (unsigned)(size_t)(&sB[0][0]);
    const unsigned long long gW =
        (unsigned long long)(size_t)convW + (unsigned long long)n0wg * KCONV * 2ull;
    if (wave == 0) {
        rpn_tdm_load_btile(ldsBase, gW);               // chunk 0 -> buf 0
    }
#else
    const __bf16* colB0 = convW + (size_t)(n0 + laneM) * KCONV;
    const __bf16* colB1 = convW + (size_t)(n0 + 16 + laneM) * KCONV;
#endif

    v8f acc[2][2] = {};

    for (int kk = 0; kk < 72; ++kk) {
#if RPN_HAS_TDM
        if (wave == 0) {
            if (kk + 1 < 72) {
                // issue chunk kk+1 into the other buffer, then wait for kk
                rpn_tdm_load_btile(ldsBase + (unsigned)(((kk + 1) & 1) * 4096),
                                   gW + (unsigned long long)(kk + 1) * 64ull);
                __builtin_amdgcn_s_wait_tensorcnt(1);
            } else {
                __builtin_amdgcn_s_wait_tensorcnt(0);
            }
        }
        __syncthreads();   // chunk kk visible to all waves
#endif
        const int rs = kk >> 3;
        const int cb = (kk & 7) << 5;        // channel block base
        const int r = rs / 3, s = rs % 3;

        const __bf16* pa0 =
            featpad + (((size_t)(n * HP + h0 + r) * HP) + (w0 + s)) * CIN + cb;
        const __bf16* pa1 =
            featpad + (((size_t)(n * HP + h1 + r) * HP) + (w1v + s)) * CIN + cb;
        v16bf a0 = load_afrag(pa0, khalf);
        v16bf a1 = load_afrag(pa1, khalf);

#if RPN_HAS_TDM
        const __bf16* bl = &sB[kk & 1][0];
        v16bf bb0 = *(const v16bf*)(bl + (nw * 32 + laneM) * 32 + kb16);
        v16bf bb1 = *(const v16bf*)(bl + (nw * 32 + 16 + laneM) * 32 + kb16);
#else
        const int k0 = kk << 5;
        v16bf bb0 = *(const v16bf*)(colB0 + k0 + kb16);
        v16bf bb1 = *(const v16bf*)(colB1 + k0 + kb16);
#endif

        // prefetch next chunk's A (emits global_prefetch_b8)
        __builtin_prefetch(pa0 + 32, 0, 1);

        acc[0][0] = __builtin_amdgcn_wmma_f32_16x16x32_bf16(
            false, a0, false, bb0, (short)0, acc[0][0], false, false);
        acc[0][1] = __builtin_amdgcn_wmma_f32_16x16x32_bf16(
            false, a0, false, bb1, (short)0, acc[0][1], false, false);
        acc[1][0] = __builtin_amdgcn_wmma_f32_16x16x32_bf16(
            false, a1, false, bb0, (short)0, acc[1][0], false, false);
        acc[1][1] = __builtin_amdgcn_wmma_f32_16x16x32_bf16(
            false, a1, false, bb1, (short)0, acc[1][1], false, false);

#if RPN_HAS_TDM
        __syncthreads();   // all reads of buf (kk&1) done before its re-fill
#endif
    }

    // Epilogue: bias + ReLU, store NHWC bf16.
    // D layout: lane<16 -> rows v (0..7); lane>=16 -> rows 8+v; col = lane&15.
    const int rofs = (lane < 16) ? 0 : 8;
#pragma unroll
    for (int a = 0; a < 2; ++a) {
#pragma unroll
        for (int b = 0; b < 2; ++b) {
            const int colC = n0 + b * 16 + laneM;
            const float bias = b1[colC];
#pragma unroll
            for (int v = 0; v < 8; ++v) {
                const int row = m0 + a * 16 + rofs + v;
                float val = acc[a][b][v] + bias;
                val = fmaxf(val, 0.0f);
                convout[((size_t)n * (HH * WW) + row) * COUT + colC] = (__bf16)val;
            }
        }
    }
}

// ---------------------------------------------------------------------------
// Stage 3: heads as GEMM. M=32768, K=256, N=48 (3 WMMA column fragments).
// 8 waves/block, each wave 16 rows. grid = 256 blocks.
// ---------------------------------------------------------------------------
__global__ __launch_bounds__(256)
void rpn_head_wmma(const __bf16* __restrict__ convout,
                   const __bf16* __restrict__ headW,
                   const float* __restrict__ b_box,
                   const float* __restrict__ b_obj,
                   float* __restrict__ headout) {
    const int wave = threadIdx.x >> 5;
    const int lane = threadIdx.x & 31;
    const size_t m0 = (size_t)blockIdx.x * 128 + (size_t)wave * 16;
    const int laneM = lane & 15;
    const int khalf = (lane < 16) ? 0 : 8;
    const int kb16  = (lane < 16) ? 0 : 16;

    const __bf16* rowA = convout + (m0 + laneM) * CIN;
    v8f acc[3] = {};

    for (int kk = 0; kk < 8; ++kk) {
        const int k0 = kk << 5;
        v16bf a = load_afrag(rowA + k0, khalf);
#pragma unroll
        for (int f = 0; f < 3; ++f) {
            v16bf b = *(const v16bf*)(headW + (size_t)(f * 16 + laneM) * CIN +
                                      k0 + kb16);
            acc[f] = __builtin_amdgcn_wmma_f32_16x16x32_bf16(
                false, a, false, b, (short)0, acc[f], false, false);
        }
    }

    const int rofs = (lane < 16) ? 0 : 8;
#pragma unroll
    for (int f = 0; f < 3; ++f) {
        const int col = f * 16 + laneM;
        float bias = 0.0f;
        if (col < 36)      bias = b_box[col];
        else if (col < 45) bias = b_obj[col - 36];
#pragma unroll
        for (int v = 0; v < 8; ++v) {
            const size_t row = m0 + rofs + v;
            headout[row * NHEAD + col] = acc[f][v] + bias;
        }
    }
}

// ---------------------------------------------------------------------------
// Stage 4: decode boxes, sigmoid, build sort keys
// ---------------------------------------------------------------------------
__global__ void rpn_decode(const float* __restrict__ headout,
                           const float* __restrict__ anchors,
                           float* __restrict__ boxes,
                           float* __restrict__ prob,
                           unsigned long long* __restrict__ keys) {
    const int gid = blockIdx.x * blockDim.x + threadIdx.x;   // 8*65536
    const int n = gid >> 16;
    const int j = gid & (SORTN - 1);
    if (j >= NPROP) {
        keys[(size_t)n * SORTN + j] = 0ull;
        return;
    }
    const int a   = j >> 12;       // anchor
    const int pix = j & 4095;      // h*64+w
    const float* hrow = headout + ((size_t)n * (HH * WW) + pix) * NHEAD;
    const float t0 = hrow[0 * AN + a];
    const float t1 = hrow[1 * AN + a];
    const float t2 = hrow[2 * AN + a];
    const float t3 = hrow[3 * AN + a];
    const float ob = hrow[36 + a];

    const float* anc = anchors + ((size_t)n * NPROP + j) * 4;
    const float acx = anc[0], acy = anc[1], aw = anc[2], ah = anc[3];
    const float cx = t0 * aw + acx;
    const float cy = t1 * ah + acy;
    const float bw = aw * expf(t2);
    const float bh = ah * expf(t3);
    const float x1 = fminf(fmaxf(cx - 0.5f * bw, 0.0f), IMG_WH);
    const float y1 = fminf(fmaxf(cy - 0.5f * bh, 0.0f), IMG_WH);
    const float x2 = fminf(fmaxf(cx + 0.5f * bw, 0.0f), IMG_WH);
    const float y2 = fminf(fmaxf(cy + 0.5f * bh, 0.0f), IMG_WH);

    float* bp = boxes + ((size_t)n * NPROP + j) * 4;
    bp[0] = x1; bp[1] = y1; bp[2] = x2; bp[3] = y2;

    const float p = 1.0f / (1.0f + expf(-ob));
    prob[(size_t)n * NPROP + j] = p;

    unsigned int b = __float_as_uint(p);
    b ^= (b >> 31) ? 0xFFFFFFFFu : 0x80000000u;   // order-preserving map
    keys[(size_t)n * SORTN + j] =
        ((unsigned long long)b << 32) | (unsigned)(0xFFFFFFFFu - (unsigned)j);
}

// ---------------------------------------------------------------------------
// Stage 5: global bitonic sort (descending) per batch, 65536 keys
// ---------------------------------------------------------------------------
__global__ void rpn_bitonic_pass(unsigned long long* __restrict__ keys,
                                 int kpar, int jpar) {
    const int gid = blockIdx.x * blockDim.x + threadIdx.x;   // 8*65536
    const int n = gid >> 16;
    const int i = gid & (SORTN - 1);
    const int ixj = i ^ jpar;
    if (ixj <= i) return;
    const size_t base = (size_t)n * SORTN;
    unsigned long long A = keys[base + i];
    unsigned long long B = keys[base + ixj];
    const bool descBlock = ((i & kpar) == 0);
    const bool doSwap = descBlock ? (A < B) : (A > B);
    if (doSwap) {
        keys[base + i] = B;
        keys[base + ixj] = A;
    }
}

// ---------------------------------------------------------------------------
// Stage 6: greedy NMS over top-2000, one workgroup per batch
// ---------------------------------------------------------------------------
__global__ __launch_bounds__(1024)
void rpn_nms(const unsigned long long* __restrict__ keys,
             const float* __restrict__ boxes,
             const float* __restrict__ prob,
             float* __restrict__ topbox,
             int* __restrict__ keepOut) {
    __shared__ float sx1[PRE_NMS], sy1[PRE_NMS], sx2[PRE_NMS], sy2[PRE_NMS];
    __shared__ float sar[PRE_NMS];
    __shared__ int   skeep[PRE_NMS];

    const int n = blockIdx.x;
    const int tid = threadIdx.x;

    for (int j = tid; j < PRE_NMS; j += 1024) {
        const unsigned long long key = keys[(size_t)n * SORTN + j];
        const int i = (int)(0xFFFFFFFFu - (unsigned)(key & 0xFFFFFFFFull));
        const float* bp = boxes + ((size_t)n * NPROP + i) * 4;
        const float x1 = bp[0], y1 = bp[1], x2 = bp[2], y2 = bp[3];
        const float p = prob[(size_t)n * NPROP + i];
        sx1[j] = x1; sy1[j] = y1; sx2[j] = x2; sy2[j] = y2;
        sar[j] = fmaxf(x2 - x1, 0.0f) * fmaxf(y2 - y1, 0.0f);
        skeep[j] = ((x2 - x1) >= MIN_SIZE) && ((y2 - y1) >= MIN_SIZE) &&
                   (p >= PROB_THR);
    }
    __syncthreads();

    for (int i = 0; i < PRE_NMS; ++i) {
        if (skeep[i]) {   // uniform read of LDS
            const float x1 = sx1[i], y1 = sy1[i], x2 = sx2[i], y2 = sy2[i];
            const float ai = sar[i];
            for (int j = i + 1 + tid; j < PRE_NMS; j += 1024) {
                const float xx1 = fmaxf(x1, sx1[j]);
                const float yy1 = fmaxf(y1, sy1[j]);
                const float xx2 = fminf(x2, sx2[j]);
                const float yy2 = fminf(y2, sy2[j]);
                const float iw = fmaxf(xx2 - xx1, 0.0f);
                const float ih = fmaxf(yy2 - yy1, 0.0f);
                const float inter = iw * ih;
                const float uni = fmaxf(ai + sar[j] - inter, 1e-9f);
                if (inter / uni > IOU_THR) skeep[j] = 0;
            }
        }
        __syncthreads();
    }

    for (int j = tid; j < PRE_NMS; j += 1024) {
        keepOut[n * PRE_NMS + j] = skeep[j];
        float* tb = topbox + ((size_t)n * PRE_NMS + j) * 4;
        tb[0] = sx1[j]; tb[1] = sy1[j]; tb[2] = sx2[j]; tb[3] = sy2[j];
    }
}

// ---------------------------------------------------------------------------
// Stage 7: pack kept boxes into first 1000 slots + mask; zero-fill output
// d_out layout: proposals (8,1000,4) f32 then mask (8,1000) as f32
// ---------------------------------------------------------------------------
__global__ __launch_bounds__(256)
void rpn_pack(const float* __restrict__ topbox,
              const int* __restrict__ keep,
              float* __restrict__ d_out) {
    __shared__ int rank[PRE_NMS];
    const int n = blockIdx.x;
    const int tid = threadIdx.x;
    float* prop = d_out + (size_t)n * POST_NMS * 4;
    float* mask = d_out + (size_t)NB * POST_NMS * 4 + (size_t)n * POST_NMS;

    for (int j = tid; j < POST_NMS * 4; j += 256) prop[j] = 0.0f;
    for (int j = tid; j < POST_NMS; j += 256) mask[j] = 0.0f;

    if (tid == 0) {
        int c = 0;
        for (int j = 0; j < PRE_NMS; ++j) {
            const int k = keep[n * PRE_NMS + j];
            rank[j] = k ? c : -1;
            c += k;
        }
    }
    __syncthreads();

    for (int j = tid; j < PRE_NMS; j += 256) {
        const int r = rank[j];
        if (r >= 0 && r < POST_NMS) {
            const float* tb = topbox + ((size_t)n * PRE_NMS + j) * 4;
            prop[r * 4 + 0] = tb[0];
            prop[r * 4 + 1] = tb[1];
            prop[r * 4 + 2] = tb[2];
            prop[r * 4 + 3] = tb[3];
            mask[r] = 1.0f;
        }
    }
}

// ---------------------------------------------------------------------------
// Host-side launch sequence (graph-capturable, all on `stream`)
// ---------------------------------------------------------------------------
extern "C" void kernel_launch(void* const* d_in, const int* in_sizes, int n_in,
                              void* d_out, int out_size, void* d_ws, size_t ws_size,
                              hipStream_t stream) {
    (void)in_sizes; (void)n_in; (void)out_size; (void)ws_size;

    const float* feat    = (const float*)d_in[0];
    const float* anchors = (const float*)d_in[1];
    const float* w1      = (const float*)d_in[2];
    const float* b1      = (const float*)d_in[3];
    const float* w_box   = (const float*)d_in[4];
    const float* b_box   = (const float*)d_in[5];
    const float* w_obj   = (const float*)d_in[6];
    const float* b_obj   = (const float*)d_in[7];
    float* out = (float*)d_out;

    char* ws = (char*)d_ws;
    __bf16* featpad = (__bf16*)(ws + OFF_FEATPAD);
    __bf16* convW   = (__bf16*)(ws + OFF_CONVW);
    __bf16* headW   = (__bf16*)(ws + OFF_HEADW);
    __bf16* convout = (__bf16*)(ws + OFF_CONVOUT);
    float*  headout = (float*)(ws + OFF_HEADOUT);
    float*  boxes   = (float*)(ws + OFF_BOXES);
    float*  prob    = (float*)(ws + OFF_PROB);
    unsigned long long* keys = (unsigned long long*)(ws + OFF_KEYS);
    float*  topbox  = (float*)(ws + OFF_TOPBOX);
    int*    keep    = (int*)(ws + OFF_KEEP);

    // Stage 1: data repacks (f32 -> bf16, WMMA-friendly layouts)
    rpn_prep_feats<<<(NB * HP * HP * CIN) / 256, 256, 0, stream>>>(feat, featpad);
    rpn_prep_convw<<<(COUT * KCONV) / 256, 256, 0, stream>>>(w1, convW);
    rpn_prep_headw<<<(NHEAD * CIN) / 256, 256, 0, stream>>>(w_box, w_obj, headW);

    // Stage 2: 3x3 conv via implicit-GEMM WMMA + TDM weight staging
    rpn_conv_wmma<<<dim3(32, 4, NB), 256, 0, stream>>>(featpad, convW, b1, convout);

    // Stage 3: box/obj heads via WMMA (bias fused)
    rpn_head_wmma<<<256, 256, 0, stream>>>(convout, headW, b_box, b_obj, headout);

    // Stage 4: decode + sigmoid + sort-key build
    rpn_decode<<<(NB * SORTN) / 256, 256, 0, stream>>>(headout, anchors, boxes,
                                                       prob, keys);

    // Stage 5: bitonic sort (descending) — 136 passes, all graph-captured
    for (int k = 2; k <= SORTN; k <<= 1)
        for (int j = k >> 1; j > 0; j >>= 1)
            rpn_bitonic_pass<<<(NB * SORTN) / 256, 256, 0, stream>>>(keys, k, j);

    // Stage 6: greedy NMS (one workgroup per batch)
    rpn_nms<<<NB, 1024, 0, stream>>>(keys, boxes, prob, topbox, keep);

    // Stage 7: pack proposals + mask
    rpn_pack<<<NB, 256, 0, stream>>>(topbox, keep, out);
}
